// GatedSelfAttention_61108794688009
// MI455X (gfx1250) — matrix operand order repaired and checked by reading, not compiled, not run on hardware
//
#include <hip/hip_runtime.h>
#include <hip/hip_bf16.h>
#include <math.h>
#include <stdint.h>

typedef __bf16 bf16_t;
typedef __attribute__((ext_vector_type(16))) __bf16 v16bf;
typedef __attribute__((ext_vector_type(8)))  __bf16 v8bf;
typedef __attribute__((ext_vector_type(2)))  __bf16 v2bf;
typedef __attribute__((ext_vector_type(8)))  float  v8f;
typedef unsigned int u32;
typedef __attribute__((ext_vector_type(4))) u32 v4u;
typedef __attribute__((ext_vector_type(8))) int v8i;
typedef __attribute__((ext_vector_type(4))) int v4i;

#define EMBED 1024
#define SEQ   2048
#define HEADS 16
#define HD    64
#define ROWS  4096   // b*s = 2*2048

#if defined(__has_builtin)
#  if __has_builtin(__builtin_amdgcn_permlane16)
#    define HAVE_PERMLANE16 1
#  endif
#  if __has_builtin(__builtin_amdgcn_cvt_pk_bf16_f32)
#    define HAVE_CVT_PK_BF16 1
#  endif
#endif

__device__ __forceinline__ float bf2f(bf16_t b) {
  unsigned short s = __builtin_bit_cast(unsigned short, b);
  unsigned u = ((unsigned)s) << 16;
  return __builtin_bit_cast(float, u);
}
__device__ __forceinline__ bf16_t f2bf(float f) {
#if defined(HAVE_CVT_PK_BF16)
  v2bf r = __builtin_amdgcn_cvt_pk_bf16_f32(f, 0.0f);
  return r[0];
#else
  unsigned u = __builtin_bit_cast(unsigned, f);
  u += 0x7FFFu + ((u >> 16) & 1u);            // round-to-nearest-even
  unsigned short s = (unsigned short)(u >> 16);
  return __builtin_bit_cast(bf16_t, s);
#endif
}
__device__ __forceinline__ v8f vzero8() {
  v8f z;
#pragma unroll
  for (int i = 0; i < 8; ++i) z[i] = 0.0f;
  return z;
}
__device__ __forceinline__ u32 lds_off(const void* p) {
  // generic pointers into LDS carry the byte offset in the low 32 bits
  return (u32)(uintptr_t)p;
}

// Max-reduction across each independent 16-lane half of the wave.
__device__ __forceinline__ float xmax16(float x) {
#if defined(HAVE_PERMLANE16)
  unsigned u = __builtin_bit_cast(unsigned, x), t;
  t = __builtin_amdgcn_permlane16(u, u, 0x67452301u, 0xEFCDAB89u, false, false); // ^1
  x = fmaxf(x, __builtin_bit_cast(float, t)); u = __builtin_bit_cast(unsigned, x);
  t = __builtin_amdgcn_permlane16(u, u, 0x54761032u, 0xDCFE98BAu, false, false); // ^2
  x = fmaxf(x, __builtin_bit_cast(float, t)); u = __builtin_bit_cast(unsigned, x);
  t = __builtin_amdgcn_permlane16(u, u, 0x32107654u, 0xBA98FEDCu, false, false); // ^4
  x = fmaxf(x, __builtin_bit_cast(float, t)); u = __builtin_bit_cast(unsigned, x);
  t = __builtin_amdgcn_permlane16(u, u, 0xFEDCBA98u, 0x76543210u, false, false); // ^8
  x = fmaxf(x, __builtin_bit_cast(float, t));
  return x;
#else
  x = fmaxf(x, __shfl_xor(x, 1));
  x = fmaxf(x, __shfl_xor(x, 2));
  x = fmaxf(x, __shfl_xor(x, 4));
  x = fmaxf(x, __shfl_xor(x, 8));
  return x;
#endif
}

// ---- Tensor Data Mover: 2D tile (tile_d1 rows x tile_d0 elems, bf16) -> LDS ----
__device__ __forceinline__ void tdm_load_2d_bf16(u32 lds_byte_off, const void* gaddr,
                                                 u32 tensor_d0, u32 tensor_d1,
                                                 u32 tile_d0, u32 tile_d1,
                                                 u32 stride0) {
  const unsigned long long ga = (unsigned long long)(uintptr_t)gaddr;
  v4u g0;
  g0[0] = 1u;                                            // count=1, user descriptor
  g0[1] = lds_byte_off;                                  // lds_addr
  g0[2] = (u32)(ga & 0xFFFFFFFFu);                       // global_addr[31:0]
  g0[3] = (u32)((ga >> 32) & 0x01FFFFFFu) | (2u << 30);  // global_addr[56:32], type=2
  v8i g1;
  g1[0] = (int)(1u << 16);                               // data_size=1 (2 bytes)
  g1[1] = (int)((tensor_d0 & 0xFFFFu) << 16);            // tensor_dim0[15:0]
  g1[2] = (int)(((tensor_d0 >> 16) & 0xFFFFu) | ((tensor_d1 & 0xFFFFu) << 16));
  g1[3] = (int)(((tensor_d1 >> 16) & 0xFFFFu) | (tile_d0 << 16));   // tile_dim0
  g1[4] = (int)(tile_d1 & 0xFFFFu);                      // tile_dim1, tile_dim2=0
  g1[5] = (int)stride0;                                  // tensor_dim0_stride[31:0]
  g1[6] = 0;
  g1[7] = 0;
  v4i g2 = {0, 0, 0, 0};
  v4i g3 = {0, 0, 0, 0};
#if defined(__clang_major__) && (__clang_major__ >= 23)
  v8i g4 = {0, 0, 0, 0, 0, 0, 0, 0};
  __builtin_amdgcn_tensor_load_to_lds(g0, g1, g2, g3, g4, 0);
#else
  __builtin_amdgcn_tensor_load_to_lds(g0, g1, g2, g3, 0);
#endif
}

// ---------------- LayerNorm (fp32 in) -> bf16 ----------------
__global__ void ln_cast_kernel(const float* __restrict__ x,
                               const float* __restrict__ gamma,
                               const float* __restrict__ beta,
                               bf16_t* __restrict__ out) {
  const int row = blockIdx.x;
  const int t = threadIdx.x;
  const float* xr = x + (size_t)row * EMBED;
  float v[4], s = 0.f, ss = 0.f;
#pragma unroll
  for (int i = 0; i < 4; ++i) {
    v[i] = xr[t + i * 256];
    s += v[i];
    ss += v[i] * v[i];
  }
  __shared__ float r1[256], r2[256];
  r1[t] = s; r2[t] = ss;
  __syncthreads();
  for (int off = 128; off > 0; off >>= 1) {
    if (t < off) { r1[t] += r1[t + off]; r2[t] += r2[t + off]; }
    __syncthreads();
  }
  const float mean = r1[0] * (1.0f / EMBED);
  const float var  = r2[0] * (1.0f / EMBED) - mean * mean;
  const float rstd = rsqrtf(var + 1e-5f);
  bf16_t* o = out + (size_t)row * EMBED;
#pragma unroll
  for (int i = 0; i < 4; ++i) {
    const int c = t + i * 256;
    o[c] = f2bf((v[i] - mean) * rstd * gamma[c] + beta[c]);
  }
}

// ---------------- fp32 -> bf16 cast ----------------
__global__ void cast_bf16_kernel(const float* __restrict__ in,
                                 bf16_t* __restrict__ out, int n) {
  const int i = blockIdx.x * blockDim.x + threadIdx.x;
  if (i < n) out[i] = f2bf(in[i]);
}

// ---------------- GEMM  Y[M,N] = X[M,K] * W[N,K]^T ------------------------
// Block: 4 waves, 256x64 tile. TDM double-buffers X(256x32) + W(64x32) tiles
// into LDS; wave 0 drives the DMA pipeline, WMMA consumes from LDS.
template <bool OUT_F32>
__global__ void gemm_xwt_kernel(const bf16_t* __restrict__ X,
                                const bf16_t* __restrict__ W,
                                bf16_t* __restrict__ Yb,
                                float* __restrict__ Yf,
                                int M, int N, int K) {
  const int lane  = threadIdx.x & 31;
  const int wv    = threadIdx.x >> 5;
  const int l16   = lane & 15;
  const int half  = lane >> 4;
  const int m_blk = blockIdx.x * 256;
  const int n_base = blockIdx.y * 64;

  __shared__ bf16_t xt[2][256 * 32];   // 32 KB
  __shared__ bf16_t wt[2][64 * 32];    //  8 KB

  v8f acc[4][4];
#pragma unroll
  for (int mt = 0; mt < 4; ++mt)
#pragma unroll
    for (int nt = 0; nt < 4; ++nt) acc[mt][nt] = vzero8();

  // Prime the pipeline: tile for k0 = 0 into buffer 0.
  if (wv == 0) {
    tdm_load_2d_bf16(lds_off(&xt[0][0]), X + (size_t)m_blk * K, K, M, 32, 256, K);
    tdm_load_2d_bf16(lds_off(&wt[0][0]), W + (size_t)n_base * K, K, N, 32, 64, K);
    __builtin_amdgcn_s_wait_tensorcnt(0);
  }
  __syncthreads();

  for (int k0 = 0; k0 < K; k0 += 32) {
    const int buf = (k0 >> 5) & 1;
    const bool has_next = (k0 + 32) < K;
    if (has_next && wv == 0) {   // kick next-tile DMA; overlaps with WMMA below
      tdm_load_2d_bf16(lds_off(&xt[buf ^ 1][0]),
                       X + (size_t)m_blk * K + (k0 + 32), K, M, 32, 256, K);
      tdm_load_2d_bf16(lds_off(&wt[buf ^ 1][0]),
                       W + (size_t)n_base * K + (k0 + 32), K, N, 32, 64, K);
    }

    v16bf a[4];
#pragma unroll
    for (int mt = 0; mt < 4; ++mt) {
      const bf16_t* p = &xt[buf][(wv * 64 + mt * 16 + l16) * 32 + half * 16];
      v8bf lo = *(const v8bf*)p;
      v8bf hi = *(const v8bf*)(p + 8);
#pragma unroll
      for (int i = 0; i < 8; ++i) { a[mt][i] = lo[i]; a[mt][i + 8] = hi[i]; }
    }
    v16bf bfr[4];
#pragma unroll
    for (int nt = 0; nt < 4; ++nt)
      bfr[nt] = *(const v16bf*)&wt[buf][(nt * 16 + l16) * 32 + half * 16];

#pragma unroll
    for (int mt = 0; mt < 4; ++mt)
#pragma unroll
      for (int nt = 0; nt < 4; ++nt)
        acc[mt][nt] = __builtin_amdgcn_wmma_f32_16x16x32_bf16(
            false, a[mt], false, bfr[nt], (short)0, acc[mt][nt], false, false);

    if (has_next && wv == 0) __builtin_amdgcn_s_wait_tensorcnt(0);
    __syncthreads();
  }

  // C layout: VGPR j -> row (half*8 + j), col l16
#pragma unroll
  for (int mt = 0; mt < 4; ++mt)
#pragma unroll
    for (int nt = 0; nt < 4; ++nt) {
#pragma unroll
      for (int j = 0; j < 8; ++j) {
        const int r = m_blk + wv * 64 + mt * 16 + half * 8 + j;
        const int c = n_base + nt * 16 + l16;
        const float v = acc[mt][nt][j];
        if (OUT_F32) Yf[(size_t)r * N + c] = v;
        else         Yb[(size_t)r * N + c] = f2bf(v);
      }
    }
}

// ---------------- RoPE in place on bf16 (interleaved pairs) ----------------
__global__ void rope_kernel(bf16_t* __restrict__ Q, int total_pairs) {
  const int pi = blockIdx.x * blockDim.x + threadIdx.x;
  if (pi >= total_pairs) return;
  const int row = pi >> 9;
  const int col = (pi & 511) * 2;
  const int p   = (col & 63) >> 1;
  const int pos = row & (SEQ - 1);
  const float freq = __powf(10000.0f, -((float)(2 * p)) * (1.0f / 64.0f));
  const float ang = (float)pos * freq;
  const float sn = __sinf(ang), cs = __cosf(ang);
  bf16_t* q = Q + (size_t)row * EMBED + col;
  const float x1 = bf2f(q[0]), x2 = bf2f(q[1]);
  q[0] = f2bf(x1 * cs - x2 * sn);
  q[1] = f2bf(x2 * cs + x1 * sn);
}

// ---------------- Flash attention: one wave per (b,h,16-query tile) ----------
// V tiles stream through per-wave double-buffered LDS via TDM (TENSORcnt),
// row-sum of P computed with an extra WMMA against an all-ones B fragment.
__global__ void attn_kernel(const bf16_t* __restrict__ Qb,
                            const bf16_t* __restrict__ Kb,
                            const bf16_t* __restrict__ Vb,
                            const bf16_t* __restrict__ Gb,
                            bf16_t* __restrict__ Ab) {
  const int lane = threadIdx.x & 31;
  const int wv   = threadIdx.x >> 5;
  const int wid  = blockIdx.x * 4 + wv;
  const int mt   = wid & 127;
  const int h    = (wid >> 7) & 15;
  const int b    = wid >> 11;
  const int l16  = lane & 15, half = lane >> 4;

  __shared__ bf16_t pbuf_s[4][16 * 32];      //  4 KB
  __shared__ bf16_t vt_s[4][2][32 * 64];     // 32 KB
  bf16_t* pbuf = pbuf_s[wv];

  const size_t rowbase = (size_t)b * SEQ;
  const int cbase = h * HD;

  // Resident Q A-frags (d = 0..31, 32..63)
  v16bf aq[2];
#pragma unroll
  for (int kt = 0; kt < 2; ++kt) {
    const bf16_t* p = Qb + (rowbase + mt * 16 + l16) * EMBED + cbase + kt * 32 + half * 16;
    v8bf lo = *(const v8bf*)p;
    v8bf hi = *(const v8bf*)(p + 8);
#pragma unroll
    for (int i = 0; i < 8; ++i) { aq[kt][i] = lo[i]; aq[kt][i + 8] = hi[i]; }
  }

  // All-ones B fragment (32x16 of bf16 1.0) for row-sum WMMA
  v16bf ones;
#pragma unroll
  for (int i = 0; i < 16; ++i) ones[i] = f2bf(1.0f);

  v8f o[4];
#pragma unroll
  for (int dt = 0; dt < 4; ++dt) o[dt] = vzero8();
  v8f lsum = vzero8();
  float mrow[8];
#pragma unroll
  for (int j = 0; j < 8; ++j) mrow[j] = -1e30f;

  const float scale = 0.125f;  // 1/sqrt(64)

  // Prime V pipeline: keys [0,32) -> buffer 0
  tdm_load_2d_bf16(lds_off(&vt_s[wv][0][0]),
                   Vb + (rowbase * EMBED) + cbase, EMBED, ROWS, 64, 32, EMBED);
  __builtin_amdgcn_s_wait_tensorcnt(0);

  for (int j0 = 0; j0 < SEQ; j0 += 32) {
    const int buf = (j0 >> 5) & 1;
    const bool has_next = (j0 + 32) < SEQ;
    if (has_next)   // next V tile DMA overlaps with S-compute + softmax
      tdm_load_2d_bf16(lds_off(&vt_s[wv][buf ^ 1][0]),
                       Vb + ((rowbase + j0 + 32) * EMBED) + cbase,
                       EMBED, ROWS, 64, 32, EMBED);

    // S = Q K^T for 32 keys (two 16-key N-tiles)
    v8f s[2];
#pragma unroll
    for (int g = 0; g < 2; ++g) {
      v16bf bk0, bk1;
      {
        const bf16_t* p = Kb + (rowbase + j0 + g * 16 + l16) * EMBED + cbase + half * 16;
        bk0 = *(const v16bf*)p;           // d = 0..31
        bk1 = *(const v16bf*)(p + 32);    // d = 32..63
        if (has_next) __builtin_prefetch(p + 32 * EMBED, 0, 1);
      }
      v8f z = vzero8();
      z = __builtin_amdgcn_wmma_f32_16x16x32_bf16(false, aq[0], false, bk0, (short)0, z, false, false);
      z = __builtin_amdgcn_wmma_f32_16x16x32_bf16(false, aq[1], false, bk1, (short)0, z, false, false);
      s[g] = z;
    }

    // Online softmax: row-max reduction over 16 lanes (halves are independent)
#pragma unroll
    for (int j = 0; j < 8; ++j) {
      float a0 = s[0][j] * scale, a1 = s[1][j] * scale;
      const float t = xmax16(fmaxf(a0, a1));
      const float nm = fmaxf(mrow[j], t);
      const float alpha = __expf(mrow[j] - nm);
      const float p0 = __expf(a0 - nm);
      const float p1 = __expf(a1 - nm);
      mrow[j] = nm;
      lsum[j] *= alpha;
#pragma unroll
      for (int dt = 0; dt < 4; ++dt) o[dt][j] *= alpha;
      const int r = half * 8 + j;
      pbuf[r * 32 + l16]      = f2bf(p0);
      pbuf[r * 32 + 16 + l16] = f2bf(p1);
    }

    // Re-load P as an A-fragment (same wave: DS ops are in order)
    v16bf ap;
#pragma unroll
    for (int i = 0; i < 8; ++i) {
      ap[i]     = pbuf[l16 * 32 + half * 16 + i];
      ap[i + 8] = pbuf[l16 * 32 + half * 16 + 8 + i];
    }

    // Row-sum via WMMA: lsum += P * ones
    lsum = __builtin_amdgcn_wmma_f32_16x16x32_bf16(
        false, ap, false, ones, (short)0, lsum, false, false);

    // O += P * V  (B[k][n] = V[key k][d n], gathered from the LDS V tile)
    const bf16_t* vtile = &vt_s[wv][buf][0];
#pragma unroll
    for (int dt = 0; dt < 4; ++dt) {
      v16bf bv;
#pragma unroll
      for (int i = 0; i < 16; ++i)
        bv[i] = vtile[(half * 16 + i) * 64 + dt * 16 + l16];
      o[dt] = __builtin_amdgcn_wmma_f32_16x16x32_bf16(
          false, ap, false, bv, (short)0, o[dt], false, false);
    }

    if (has_next) __builtin_amdgcn_s_wait_tensorcnt(0);
  }

  // Epilogue: normalize, gate with sigmoid(G), emit bf16
#pragma unroll
  for (int dt = 0; dt < 4; ++dt)
#pragma unroll
    for (int j = 0; j < 8; ++j) {
      const size_t r = rowbase + mt * 16 + half * 8 + j;
      const int c = cbase + dt * 16 + l16;
      const float g = bf2f(Gb[r * EMBED + c]);
      const float gate = 1.0f / (1.0f + __expf(-g));
      Ab[r * EMBED + c] = f2bf((o[dt][j] / lsum[j]) * gate);
    }
}

// ---------------- host-side orchestration ----------------
extern "C" void kernel_launch(void* const* d_in, const int* in_sizes, int n_in,
                              void* d_out, int out_size, void* d_ws, size_t ws_size,
                              hipStream_t stream) {
  const float* s_i  = (const float*)d_in[0];
  const float* w_q  = (const float*)d_in[1];
  const float* w_k  = (const float*)d_in[2];
  const float* w_v  = (const float*)d_in[3];
  const float* w_g  = (const float*)d_in[4];
  const float* w_o  = (const float*)d_in[5];
  const float* ln_g = (const float*)d_in[6];
  const float* ln_b = (const float*)d_in[7];
  float* out = (float*)d_out;

  char* ws = (char*)d_ws;
  size_t off = 0;
  const size_t actN = (size_t)ROWS * EMBED;
  const size_t wN   = (size_t)EMBED * EMBED;

  bf16_t* Xb  = (bf16_t*)(ws + off); off += actN * 2;
  bf16_t* Wqb = (bf16_t*)(ws + off); off += wN * 2;
  bf16_t* Wkb = (bf16_t*)(ws + off); off += wN * 2;
  bf16_t* Wvb = (bf16_t*)(ws + off); off += wN * 2;
  bf16_t* Wgb = (bf16_t*)(ws + off); off += wN * 2;
  bf16_t* Wob = (bf16_t*)(ws + off); off += wN * 2;
  bf16_t* Qb  = (bf16_t*)(ws + off); off += actN * 2;
  bf16_t* Kb  = (bf16_t*)(ws + off); off += actN * 2;
  bf16_t* Vb  = (bf16_t*)(ws + off); off += actN * 2;
  bf16_t* Gb  = (bf16_t*)(ws + off); off += actN * 2;
  bf16_t* Ab  = (bf16_t*)(ws + off); off += actN * 2;

  ln_cast_kernel<<<ROWS, 256, 0, stream>>>(s_i, ln_g, ln_b, Xb);

  const int wn = (int)wN;
  cast_bf16_kernel<<<(wn + 255) / 256, 256, 0, stream>>>(w_q, Wqb, wn);
  cast_bf16_kernel<<<(wn + 255) / 256, 256, 0, stream>>>(w_k, Wkb, wn);
  cast_bf16_kernel<<<(wn + 255) / 256, 256, 0, stream>>>(w_v, Wvb, wn);
  cast_bf16_kernel<<<(wn + 255) / 256, 256, 0, stream>>>(w_g, Wgb, wn);
  cast_bf16_kernel<<<(wn + 255) / 256, 256, 0, stream>>>(w_o, Wob, wn);

  dim3 ggrid(ROWS / 256, EMBED / 64);
  gemm_xwt_kernel<false><<<ggrid, 128, 0, stream>>>(Xb, Wqb, Qb, nullptr, ROWS, EMBED, EMBED);
  gemm_xwt_kernel<false><<<ggrid, 128, 0, stream>>>(Xb, Wkb, Kb, nullptr, ROWS, EMBED, EMBED);
  gemm_xwt_kernel<false><<<ggrid, 128, 0, stream>>>(Xb, Wvb, Vb, nullptr, ROWS, EMBED, EMBED);
  gemm_xwt_kernel<false><<<ggrid, 128, 0, stream>>>(Xb, Wgb, Gb, nullptr, ROWS, EMBED, EMBED);

  const int pairs = (int)(actN / 2);
  rope_kernel<<<(pairs + 255) / 256, 256, 0, stream>>>(Qb, pairs);
  rope_kernel<<<(pairs + 255) / 256, 256, 0, stream>>>(Kb, pairs);

  attn_kernel<<<(2 * HEADS * (SEQ / 16)) / 4, 128, 0, stream>>>(Qb, Kb, Vb, Gb, Ab);

  gemm_xwt_kernel<true><<<ggrid, 128, 0, stream>>>(Ab, Wob, nullptr, out, ROWS, EMBED, EMBED);
}
